// DQResNet_62835371540988
// MI455X (gfx1250) — matrix operand order repaired and checked by reading, not compiled, not run on hardware
//
#include <hip/hip_runtime.h>
#include <hip/hip_bf16.h>

typedef __attribute__((ext_vector_type(16))) _Float16 v16h;
typedef __attribute__((ext_vector_type(8)))  float    v8f;
typedef __attribute__((ext_vector_type(4))) unsigned int u32x4;
typedef __attribute__((ext_vector_type(8))) int          i32x8;
typedef __attribute__((ext_vector_type(4))) int          i32x4;

#define FINF 3.402823466e+38f

// ---------------------------------------------------------------------------
// Conv as implicit GEMM on WMMA f16 (16x16x32), fully shape-specialized.
// Input image staged in LDS (TDM tensor_load_to_lds for the pre-quantized f16
// path, manual cvt for the f32 shortcut path). Activations are halo-padded in
// global memory so the 3x3 gather from LDS is guard-free. Wave computes
// 2 M-tiles x (COUT/16) N-tiles; B loaded as 2x uint4 from K-padded rows.
// BN (+ReLU) fused in epilogue.
// ---------------------------------------------------------------------------
template<int CIN, int COUT, int HIN, int STRIDE, int KS, int QUANT>
__global__ __launch_bounds__(256) void conv_wmma(const _Float16* __restrict__ qin,
                                                 const float* __restrict__ fin,
                                                 float* __restrict__ out,
                                                 const _Float16* __restrict__ wqbase,
                                                 long wq_stride,
                                                 const int* __restrict__ bitrow,
                                                 const float* __restrict__ gam,
                                                 const float* __restrict__ bet,
                                                 const float* __restrict__ mea,
                                                 const float* __restrict__ var,
                                                 int relu) {
  constexpr int HOUT = HIN / STRIDE;
  constexpr int WOUT = HOUT;
  constexpr int M    = HOUT * WOUT;          // multiple of 32 for all configs
  constexpr int K0   = CIN * KS * KS;
  constexpr int KC   = (K0 + 31) / 32;
  constexpr int KP   = KC * 32;              // padded K (weight row stride)
  constexpr int TN   = COUT / 16;            // N-tiles per wave (1,2,4)
  constexpr int T    = M / 32;               // wave-tiles (2 M-tiles each)
  constexpr int HP   = (KS == 3) ? HIN + 2 : HIN;   // halo-padded input extent
  constexpr int SP   = (CIN * HP * HP + 7) & ~7;    // per-image stride (halves)

  __shared__ _Float16 sIn[SP];

  const int n    = blockIdx.x;
  const int wave = threadIdx.x >> 5;
  const int lane = threadIdx.x & 31;
  const int t    = blockIdx.y * 8 + wave;
  const int hl   = lane & 15;
  const int hi   = lane >> 4;

  // ---- stage the whole (padded) input image into LDS -----------------------
  if (QUANT) {
#if __has_builtin(__builtin_amdgcn_tensor_load_to_lds)
    if (wave == 0) {
      // TDM descriptor: 1 x SP tile of 2-byte elements, global -> LDS.
      const unsigned lds_off = (unsigned)(size_t)(void*)&sIn[0];
      const unsigned long long ga = (unsigned long long)(size_t)(qin + (size_t)n * SP);
      u32x4 g0 = {};
      i32x8 g1 = {};
      i32x4 g2 = {}, g3 = {};
      i32x8 g4 = {};
      g0[0] = 1u;                                            // count = 1 (valid D#)
      g0[1] = lds_off;                                       // lds_addr
      g0[2] = (unsigned)(ga & 0xffffffffull);                // global_addr[31:0]
      g0[3] = (unsigned)((ga >> 32) & 0x01ffffffull)         // global_addr[56:32]
              | (2u << 30);                                  // type = 2 (image)
      g1[0] = (int)(1u << 16);                               // data_size = 2 bytes
      g1[1] = (int)(((unsigned)SP & 0xffffu) << 16);         // tensor_dim0 lo
      g1[2] = (int)((((unsigned)SP >> 16) & 0xffffu)         // tensor_dim0 hi
                    | (1u << 16));                           // tensor_dim1 = 1
      g1[3] = (int)((unsigned)SP << 16);                     // tile_dim0 = SP
      g1[4] = 1;                                             // tile_dim1 = 1
      g1[5] = SP;                                            // tensor_dim0_stride
      __builtin_amdgcn_tensor_load_to_lds(g0, g1, g2, g3, g4, 0);
      __builtin_amdgcn_s_wait_tensorcnt(0);
    }
#else
    const uint4* src = (const uint4*)(qin + (size_t)n * SP);
    uint4* dst = (uint4*)sIn;
    for (int i = threadIdx.x; i < SP / 8; i += 256) dst[i] = src[i];
#endif
  } else {
    const float* finN = fin + (size_t)n * CIN * HIN * HIN;
    for (int i = threadIdx.x; i < CIN * HIN * HIN; i += 256) sIn[i] = (_Float16)finN[i];
  }
  __syncthreads();

  if (t < T) {                               // uniform per wave -> EXEC all-ones
    const int bit = (QUANT && bitrow) ? bitrow[n] : 0;
    const _Float16* wq = wqbase + (size_t)bit * wq_stride;

    v8f acc[2][TN] = {};

    for (int kc = 0; kc < KC; ++kc) {
      const int kbase = kc * 32;

      // B fragments: 32x16 f16; lanes 0-15 K 0..15, lanes 16-31 K 16..31.
      v16h B[TN];
#pragma unroll
      for (int nt = 0; nt < TN; ++nt) {
        const uint4* wrow = reinterpret_cast<const uint4*>(wq + (size_t)(nt * 16 + hl) * KP);
        struct U32B { uint4 a, b; } raw;
        raw.a = wrow[kc * 4 + hi * 2 + 0];
        raw.b = wrow[kc * 4 + hi * 2 + 1];
        B[nt] = __builtin_bit_cast(v16h, raw);
      }
      if (kc + 1 < KC)
        __builtin_prefetch(wq + (size_t)hl * KP + (kc + 1) * 32, 0, 0);  // global_prefetch_b8

      // 2 M-tiles reuse the B fragments; A gathered guard-free from LDS halo.
#pragma unroll
      for (int mt = 0; mt < 2; ++mt) {
        const int mrow = t * 32 + mt * 16 + hl;    // A row (spatial index)
        const int py = mrow / WOUT;                // constant divisor -> folds
        const int px = mrow - py * WOUT;
        v16h A;
#pragma unroll
        for (int h = 0; h < 16; ++h) {
          // A 16x32 layout: lanes 0-15 K 0..15; lanes 16-31 K 16..31
          const int ka = kbase + (hi << 4) + h;
          _Float16 va = (_Float16)0.f;
          if (ka < K0) {
            if (KS == 1) {
              va = sIn[ka * HIN * HIN + (py * STRIDE) * HIN + px * STRIDE];
            } else {
              const int ci = ka / 9;
              const int r = ka - ci * 9;
              const int ky = r / 3, kx = r - ky * 3;
              va = sIn[ci * HP * HP + (py * STRIDE + ky) * HP + (px * STRIDE + kx)];
            }
          }
          A[h] = va;
        }
#pragma unroll
        for (int nt = 0; nt < TN; ++nt)
          acc[mt][nt] = __builtin_amdgcn_wmma_f32_16x16x32_f16(
              false, A, false, B[nt], (short)0, acc[mt][nt], false, false);
      }
    }

    // epilogue: BN (+ReLU); D layout: lane = ncol + 16*(m>=8), vgpr r -> m = r + 8*hi
    float* outN = out + (size_t)n * COUT * M;
#pragma unroll
    for (int nt = 0; nt < TN; ++nt) {
      const int ncol = nt * 16 + hl;
      const float iv = gam[ncol] * rsqrtf(var[ncol] + 1e-5f);
      const float bb = bet[ncol] - mea[ncol] * iv;
#pragma unroll
      for (int mt = 0; mt < 2; ++mt) {
#pragma unroll
        for (int r = 0; r < 8; ++r) {
          const int mr = t * 32 + mt * 16 + (hi << 3) + r;
          float vv = acc[mt][nt][r] * iv + bb;
          if (relu) vv = fmaxf(vv, 0.f);
          outN[(size_t)ncol * M + mr] = vv;
        }
      }
    }
  }
}

// ---------------------------------------------------------------------------
// Per-bit masked min/max of an activation buffer (two-pass reduction).
// ---------------------------------------------------------------------------
__global__ __launch_bounds__(256) void act_stats_partial(const float* __restrict__ act,
                                                         const int* __restrict__ bitrow,
                                                         long perImg, long total,
                                                         float* __restrict__ partial) {
  float mn0 = FINF, mn1 = FINF, mn2 = FINF, mx0 = -FINF, mx1 = -FINF, mx2 = -FINF;
  for (long i = (long)blockIdx.x * blockDim.x + threadIdx.x; i < total;
       i += (long)gridDim.x * blockDim.x) {
    const int nimg = (int)(i / perImg);
    const int b = bitrow[nimg];
    const float v = act[i];
    if (b == 0)      { mn0 = fminf(mn0, v); mx0 = fmaxf(mx0, v); }
    else if (b == 1) { mn1 = fminf(mn1, v); mx1 = fmaxf(mx1, v); }
    else             { mn2 = fminf(mn2, v); mx2 = fmaxf(mx2, v); }
  }
  __shared__ float s[6][256];
  const int tid = threadIdx.x;
  s[0][tid] = mn0; s[1][tid] = mn1; s[2][tid] = mn2;
  s[3][tid] = mx0; s[4][tid] = mx1; s[5][tid] = mx2;
  __syncthreads();
  for (int st = 128; st > 0; st >>= 1) {
    if (tid < st) {
      for (int j = 0; j < 3; ++j) {
        s[j][tid]     = fminf(s[j][tid],     s[j][tid + st]);
        s[j + 3][tid] = fmaxf(s[j + 3][tid], s[j + 3][tid + st]);
      }
    }
    __syncthreads();
  }
  if (tid == 0)
    for (int j = 0; j < 6; ++j) partial[blockIdx.x * 6 + j] = s[j][0];
}

__global__ void act_stats_final(const float* __restrict__ partial, int nPart,
                                float* __restrict__ stats) {
  if (threadIdx.x == 0 && blockIdx.x == 0) {
    const float lv[3] = {3.f, 15.f, 255.f};
    for (int b = 0; b < 3; ++b) {
      float mn = FINF, mx = -FINF;
      for (int p = 0; p < nPart; ++p) {
        mn = fminf(mn, partial[p * 6 + b]);
        mx = fmaxf(mx, partial[p * 6 + b + 3]);
      }
      if (!(mn <= mx)) { mn = 0.f; mx = 0.f; }   // no image with this bit
      stats[b * 2]     = mn;
      stats[b * 2 + 1] = fmaxf((mx - mn) / lv[b], 1e-8f);
    }
  }
}

// ---------------------------------------------------------------------------
// Quantize activations once into f16, writing a zero halo border:
// layout [n][C][Hn+2][Hn+2], per-image stride SPr (aligned to 8 halves).
// ---------------------------------------------------------------------------
__global__ __launch_bounds__(256) void quant_act_pad(const float* __restrict__ act,
                                                     _Float16* __restrict__ qact,
                                                     const int* __restrict__ bitrow,
                                                     const float* __restrict__ stats,
                                                     int C, int Hn, int SPr, long total) {
  const int HP = Hn + 2;
  const int hp2 = HP * HP;
  long i  = (long)blockIdx.x * blockDim.x + threadIdx.x;
  const long st = (long)gridDim.x * blockDim.x;
  for (; i < total; i += st) {
    const int nimg = (int)(i / SPr);
    const int r = (int)(i - (long)nimg * SPr);
    const int ci = r / hp2;
    const int p = r - ci * hp2;
    const int y = p / HP, xx = p - y * HP;
    float outv = 0.f;
    if (ci < C && (unsigned)(y - 1) < (unsigned)Hn && (unsigned)(xx - 1) < (unsigned)Hn) {
      const int b = bitrow[nimg];
      const float mn = stats[b * 2], sc = stats[b * 2 + 1];
      const float v = act[(size_t)nimg * C * Hn * Hn + (size_t)ci * Hn * Hn
                          + (size_t)(y - 1) * Hn + (xx - 1)];
      outv = rintf((v - mn) / sc) * sc + mn;
    }
    qact[i] = (_Float16)outv;
  }
}

// ---------------------------------------------------------------------------
// Weight quantization -> 3 f16 copies, rows padded to KP (zeros in the pad).
// ---------------------------------------------------------------------------
__global__ __launch_bounds__(256) void quant_weights(const float* __restrict__ w, int rows,
                                                     int K0_, int KP_, _Float16* __restrict__ qw) {
  __shared__ float smn[256], smx[256];
  const int tid = threadIdx.x;
  const int n = rows * K0_;
  float mn = FINF, mx = -FINF;
  for (int i = tid; i < n; i += 256) { const float v = w[i]; mn = fminf(mn, v); mx = fmaxf(mx, v); }
  smn[tid] = mn; smx[tid] = mx; __syncthreads();
  for (int st = 128; st > 0; st >>= 1) {
    if (tid < st) { smn[tid] = fminf(smn[tid], smn[tid + st]); smx[tid] = fmaxf(smx[tid], smx[tid + st]); }
    __syncthreads();
  }
  const float gmn = smn[0], gmx = smx[0];
  const float lv[3] = {3.f, 15.f, 255.f};
  const int np = rows * KP_;
  for (int b = 0; b < 3; ++b) {
    const float sc2 = fmaxf((gmx - gmn) / lv[b], 1e-8f);
    const float is = 1.f / sc2;
    for (int i = tid; i < np; i += 256) {
      const int co = i / KP_, k = i - co * KP_;
      qw[(size_t)b * np + i] =
          (k < K0_) ? (_Float16)(rintf((w[co * K0_ + k] - gmn) * is) * sc2 + gmn) : (_Float16)0.f;
    }
  }
}

__global__ void cvt_f16_pad(const float* __restrict__ w, int K0_, int KP_, int rows,
                            _Float16* __restrict__ o) {
  const int i = blockIdx.x * blockDim.x + threadIdx.x;
  if (i < rows * KP_) {
    const int co = i / KP_, k = i - co * KP_;
    o[i] = (k < K0_) ? (_Float16)w[co * K0_ + k] : (_Float16)0.f;
  }
}

__global__ void add_relu_k(const float* __restrict__ a, const float* __restrict__ b,
                           float* __restrict__ o, long n) {
  long i  = (long)blockIdx.x * blockDim.x + threadIdx.x;
  long st = (long)gridDim.x * blockDim.x;
  for (; i < n; i += st) o[i] = fmaxf(a[i] + b[i], 0.f);
}

// ---------------------------------------------------------------------------
// Controller: conv(3->16,s2) relu, conv(16->32,s2) relu, mean, 10x argmax head.
// ---------------------------------------------------------------------------
__global__ __launch_bounds__(256) void controller_k(const float* __restrict__ x,
                                                    const float* __restrict__ w1, const float* __restrict__ b1,
                                                    const float* __restrict__ w2, const float* __restrict__ b2,
                                                    const float* __restrict__ hw, const float* __restrict__ hb,
                                                    float* __restrict__ bits_f, int* __restrict__ bitsI) {
  const int n = blockIdx.x;
  __shared__ float f1[16 * 16 * 16];
  __shared__ float f2[32 * 8 * 8];
  __shared__ float feat[32];
  const float* xn = x + (size_t)n * 3 * 32 * 32;
  for (int i = threadIdx.x; i < 16 * 256; i += 256) {
    const int c = i >> 8, p = i & 255, y = p >> 4, xx = p & 15;
    float sacc = b1[c];
    for (int ci = 0; ci < 3; ++ci)
      for (int ky = 0; ky < 3; ++ky) {
        const int iy = 2 * y + ky - 1;
        if ((unsigned)iy < 32u)
          for (int kx = 0; kx < 3; ++kx) {
            const int ix = 2 * xx + kx - 1;
            if ((unsigned)ix < 32u)
              sacc += xn[ci * 1024 + iy * 32 + ix] * w1[((c * 3 + ci) * 3 + ky) * 3 + kx];
          }
      }
    f1[i] = fmaxf(sacc, 0.f);
  }
  __syncthreads();
  for (int i = threadIdx.x; i < 32 * 64; i += 256) {
    const int c = i >> 6, p = i & 63, y = p >> 3, xx = p & 7;
    float sacc = b2[c];
    for (int ci = 0; ci < 16; ++ci)
      for (int ky = 0; ky < 3; ++ky) {
        const int iy = 2 * y + ky - 1;
        if ((unsigned)iy < 16u)
          for (int kx = 0; kx < 3; ++kx) {
            const int ix = 2 * xx + kx - 1;
            if ((unsigned)ix < 16u)
              sacc += f1[ci * 256 + iy * 16 + ix] * w2[((c * 16 + ci) * 3 + ky) * 3 + kx];
          }
      }
    f2[i] = fmaxf(sacc, 0.f);
  }
  __syncthreads();
  if (threadIdx.x < 32) {
    float sacc = 0.f;
    for (int p = 0; p < 64; ++p) sacc += f2[threadIdx.x * 64 + p];
    feat[threadIdx.x] = sacc * (1.f / 64.f);
  }
  __syncthreads();
  if (threadIdx.x < 10) {
    const int h = threadIdx.x;
    float best = -FINF; int bi = 0;
    for (int o = 0; o < 3; ++o) {
      float lg = hb[h * 3 + o];
      for (int f = 0; f < 32; ++f) lg += feat[f] * hw[(h * 3 + o) * 32 + f];
      if (lg > best) { best = lg; bi = o; }      // first-max like jnp.argmax
    }
    bits_f[h * 1024 + n] = (bi == 0) ? 2.f : ((bi == 1) ? 4.f : 8.f);
    bitsI[h * 1024 + n] = bi;
  }
}

// ---------------------------------------------------------------------------
// Global avg pool (8x8, 64ch) + 64->10 linear.
// ---------------------------------------------------------------------------
__global__ __launch_bounds__(64) void head_k(const float* __restrict__ act,
                                             const float* __restrict__ lw, const float* __restrict__ lb,
                                             float* __restrict__ out) {
  const int n = blockIdx.x;
  __shared__ float pooled[64];
  const int c = threadIdx.x;
  const float* p = act + (size_t)n * 64 * 64 + (size_t)c * 64;
  float sacc = 0.f;
  for (int i = 0; i < 64; ++i) sacc += p[i];
  pooled[c] = sacc * (1.f / 64.f);
  __syncthreads();
  if (c < 10) {
    float s2 = lb[c];
    for (int k = 0; k < 64; ++k) s2 += pooled[k] * lw[c * 64 + k];
    out[(size_t)n * 10 + c] = s2;
  }
}

// ---------------------------------------------------------------------------
// Host orchestration
// ---------------------------------------------------------------------------
static void conv_dispatch(int cfg, hipStream_t st, const _Float16* qin, const float* fin,
                          float* out, const _Float16* wq, long wqs, const int* bitrow,
                          const float* gam, const float* bet, const float* mea, const float* var,
                          int relu) {
#define LCONV(CI, CO, HI, STR, KSZ, Q)                                                   \
  {                                                                                      \
    constexpr int M_ = (HI / STR) * (HI / STR);                                          \
    constexpr int T_ = M_ / 32;                                                          \
    dim3 gd(1024, (unsigned)((T_ + 7) / 8));                                             \
    conv_wmma<CI, CO, HI, STR, KSZ, Q>                                                   \
        <<<gd, 256, 0, st>>>(qin, fin, out, wq, wqs, bitrow, gam, bet, mea, var, relu);  \
  }
  switch (cfg) {
    case 0: LCONV(3, 16, 32, 1, 3, 1); break;   // main conv1
    case 1: LCONV(16, 16, 32, 1, 3, 1); break;  // blocks 0-2 conv1/conv2
    case 2: LCONV(16, 32, 32, 2, 3, 1); break;  // block 3 conv1
    case 3: LCONV(32, 32, 16, 1, 3, 1); break;  // block 3 conv2, blocks 4-5
    case 4: LCONV(32, 64, 16, 2, 3, 1); break;  // block 6 conv1
    case 5: LCONV(64, 64, 8, 1, 3, 1); break;   // block 6 conv2, blocks 7-8
    case 6: LCONV(16, 32, 32, 2, 1, 0); break;  // block 3 shortcut
    case 7: LCONV(32, 64, 16, 2, 1, 0); break;  // block 6 shortcut
  }
#undef LCONV
}

extern "C" void kernel_launch(void* const* d_in, const int* in_sizes, int n_in,
                              void* d_out, int out_size, void* d_ws, size_t ws_size,
                              hipStream_t stream) {
  (void)in_sizes; (void)n_in; (void)out_size; (void)ws_size;
  auto F = [&](int i) { return (const float*)d_in[i]; };
  const float* x = F(0);

  static const int CIN[9]  = {16, 16, 16, 16, 32, 32, 32, 64, 64};
  static const int COUT[9] = {16, 16, 16, 32, 32, 32, 64, 64, 64};
  static const int STR[9]  = { 1,  1,  1,  2,  1,  1,  2,  1,  1};
  static const int C1CFG[9] = {1, 1, 1, 2, 3, 3, 4, 5, 5};
  static const int C2CFG[9] = {1, 1, 1, 3, 3, 3, 5, 5, 5};
  auto KPof = [](int k0) { return ((k0 + 31) / 32) * 32; };
  auto SPof = [](int C, int Hn) { int v = C * (Hn + 2) * (Hn + 2); return (v + 7) & ~7; };

  // d_in leaf index tables (setup_inputs dict flatten order)
  int w1i[9], bn1i[9], w2i[9], bn2i[9], scwi[9], scbni[9];
  {
    int base = 6;
    for (int i = 0; i < 9; ++i) {
      const bool sc = (STR[i] != 1) || (CIN[i] != COUT[i]);
      w1i[i] = base; bn1i[i] = base + 1; w2i[i] = base + 5; bn2i[i] = base + 6;
      scwi[i] = sc ? base + 10 : -1; scbni[i] = sc ? base + 11 : -1;
      base += sc ? 15 : 10;
    }
  }
  const int LINW = 106, LINB = 107, CW1 = 108, CB1 = 109, CW2 = 110, CB2 = 111,
            CHW = 112, CHB = 113;

  // quantized-weight arena offsets (in halves, K-padded rows, x3 bit copies)
  size_t qwTot = 0, qwConv1, qw1o[9], qw2o[9];
  auto qa = [&](size_t nels) { size_t o = qwTot; qwTot += nels * 3; return o; };
  qwConv1 = qa(16 * KPof(3 * 9));
  for (int i = 0; i < 9; ++i) {
    qw1o[i] = qa((size_t)COUT[i] * KPof(CIN[i] * 9));
    qw2o[i] = qa((size_t)COUT[i] * KPof(COUT[i] * 9));
  }

  // workspace layout
  char* ws = (char*)d_ws;
  size_t off = 0;
  auto wa = [&](size_t bytes) { size_t o = (off + 255) & ~(size_t)255; off = o + bytes; return o; };
  int*      bitsI   = (int*)(ws + wa(10 * 1024 * sizeof(int)));
  float*    partial = (float*)(ws + wa(256 * 6 * sizeof(float)));
  float*    stats   = (float*)(ws + wa(8 * sizeof(float)));
  _Float16* qwA     = (_Float16*)(ws + wa(qwTot * sizeof(_Float16)));
  _Float16* scw3    = (_Float16*)(ws + wa((size_t)32 * 32 * sizeof(_Float16)));
  _Float16* scw6    = (_Float16*)(ws + wa((size_t)64 * 32 * sizeof(_Float16)));
  _Float16* qact    = (_Float16*)(ws + wa((size_t)1024 * 18496 * sizeof(_Float16)));
  const size_t bufBytes = (size_t)1024 * 16 * 32 * 32 * sizeof(float);
  float* bufA = (float*)(ws + wa(bufBytes));
  float* bufB = (float*)(ws + wa(bufBytes));
  float* bufC = (float*)(ws + wa(bufBytes));

  float* logits = (float*)d_out;
  float* bits_f = logits + 1024 * 10;

  // 1) controller -> bits (d_out tail) + bit codes (ws)
  controller_k<<<1024, 256, 0, stream>>>(x, F(CW1), F(CB1), F(CW2), F(CB2), F(CHW), F(CHB),
                                         bits_f, bitsI);

  // 2) weight quantization (3 bit levels, padded rows) + f16 shortcut weights
  quant_weights<<<1, 256, 0, stream>>>(F(1), 16, 27, KPof(27), qwA + qwConv1);
  for (int i = 0; i < 9; ++i) {
    quant_weights<<<1, 256, 0, stream>>>(F(w1i[i]), COUT[i], CIN[i] * 9, KPof(CIN[i] * 9),
                                         qwA + qw1o[i]);
    quant_weights<<<1, 256, 0, stream>>>(F(w2i[i]), COUT[i], COUT[i] * 9, KPof(COUT[i] * 9),
                                         qwA + qw2o[i]);
  }
  cvt_f16_pad<<<4, 256, 0, stream>>>(F(scwi[3]), 16, 32, 32, scw3);
  cvt_f16_pad<<<8, 256, 0, stream>>>(F(scwi[6]), 32, 32, 64, scw6);

  // helper: per-bit stats -> halo-padded f16 quantized activations
  auto stats_quant = [&](const float* act, const int* brow, int C, int Hn) {
    const long perImg = (long)C * Hn * Hn;
    act_stats_partial<<<256, 256, 0, stream>>>(act, brow, perImg, perImg * 1024, partial);
    act_stats_final<<<1, 1, 0, stream>>>(partial, 256, stats);
    const int SPr = SPof(C, Hn);
    quant_act_pad<<<2048, 256, 0, stream>>>(act, qact, brow, stats, C, Hn, SPr,
                                            (long)SPr * 1024);
  };

  // 3) main conv1: x -> bufA (bits row 0)
  stats_quant(x, bitsI, 3, 32);
  conv_dispatch(0, stream, qact, nullptr, bufA, qwA + qwConv1, 16L * KPof(27), bitsI,
                F(2), F(3), F(4), F(5), /*relu=*/1);

  // 4) residual blocks
  int H = 32, cin = 16;
  for (int i = 0; i < 9; ++i) {
    const int cout = COUT[i], s = STR[i], Hout = H / s;
    const int* brow = bitsI + (i + 1) * 1024;

    stats_quant(bufA, brow, cin, H);
    conv_dispatch(C1CFG[i], stream, qact, nullptr, bufB, qwA + qw1o[i],
                  (long)cout * KPof(cin * 9), brow,
                  F(bn1i[i]), F(bn1i[i] + 1), F(bn1i[i] + 2), F(bn1i[i] + 3), /*relu=*/1);

    stats_quant(bufB, brow, cout, Hout);
    conv_dispatch(C2CFG[i], stream, qact, nullptr, bufC, qwA + qw2o[i],
                  (long)cout * KPof(cout * 9), brow,
                  F(bn2i[i]), F(bn2i[i] + 1), F(bn2i[i] + 2), F(bn2i[i] + 3), /*relu=*/0);

    const long outN = 1024L * cout * Hout * Hout;
    int g = (int)((outN + 255) / 256); if (g > 4096) g = 4096;
    if (scwi[i] >= 0) {
      conv_dispatch((i == 3) ? 6 : 7, stream, nullptr, bufA, bufB,
                    (i == 3) ? scw3 : scw6, 0, nullptr,
                    F(scbni[i]), F(scbni[i] + 1), F(scbni[i] + 2), F(scbni[i] + 3), /*relu=*/0);
      add_relu_k<<<g, 256, 0, stream>>>(bufC, bufB, bufA, outN);
    } else {
      add_relu_k<<<g, 256, 0, stream>>>(bufC, bufA, bufA, outN);
    }
    H = Hout; cin = cout;
  }

  // 5) pool + linear -> logits
  head_k<<<1024, 64, 0, stream>>>(bufA, F(LINW), F(LINB), logits);
}
